// RNN_14035953123610
// MI455X (gfx1250) — compile-verified, matrix operand order checked by reading, not run on hardware
//
#include <hip/hip_runtime.h>
#include <cstddef>
#include <cstdint>

// ---------------------------------------------------------------------------
// LSTM language model forward (log-likelihoods), bf16 WMMA GEMMs on gfx1250.
// V=32000, E=H=1024, N=32 (batch), T=64. Output: (N, T-1) f32.
// Round 3: occupancy fixes grounded in bandwidth-delay product:
//   - gates GEMM: split-K=2 in one launch (128 wave-units, partial sums
//     reduced in the LSTM cell kernel)
//   - emit GEMM: 32-wide strips -> 1000 waves (B still read exactly once)
// ---------------------------------------------------------------------------

typedef __bf16 bf16_t;
typedef __attribute__((ext_vector_type(8)))  __bf16 v8bf;
typedef __attribute__((ext_vector_type(16))) __bf16 v16bf;
typedef __attribute__((ext_vector_type(8)))  float  v8f;

static constexpr int V_ = 32000;
static constexpr int E_ = 1024;
static constexpr int H_ = 1024;
static constexpr int N_ = 32;
static constexpr int T_ = 64;
static constexpr int G_ = 4 * H_;       // 4096 gate columns
static constexpr int KCAT_ = E_ + H_;   // 2048 (x | h concatenated)

union Frag16 { v16bf v; v8bf h[2]; };

// One lane's A or B fragment for v_wmma_f32_16x16x32_bf16 (ISA 7.12.2):
// lane L<16 holds row/col L with K blocks [0..7] (v0..v3) and [16..23]
// (v4..v7); lanes 16..31 hold the same row index with K blocks [8..15] and
// [24..31]. Both chunks are contiguous, 16-byte-aligned b128 loads.
__device__ __forceinline__ v16bf load_frag(const bf16_t* __restrict__ row,
                                           int k, int half) {
  Frag16 f;
  f.h[0] = *reinterpret_cast<const v8bf*>(row + k + half * 8);
  f.h[1] = *reinterpret_cast<const v8bf*>(row + k + 16 + half * 8);
  return f.v;
}

template <int TILES>
struct Frags {
  v16bf a0, a1, b[TILES];
};

#define WMMA_BF16(c, a, b)                                                  \
  c = __builtin_amdgcn_wmma_f32_16x16x32_bf16(false, a, false, b, (short)0, \
                                              c, false, false)

// D = A (32 x K, bf16, stride lda) @ B^T (ncols x K, bf16, stride ldb) + bias.
// Each wave-unit owns a 32 x (TILES*16) strip and 1/KSPLIT of the K range;
// partial results go to out + ks*part_stride. Every B element is loaded
// exactly once per GEMM (strips own disjoint columns, K-splits disjoint K).
template <int TILES, int KSPLIT>
__global__ void gemm_bf16_wmma(const bf16_t* __restrict__ A, int lda,
                               const bf16_t* __restrict__ B, int ldb,
                               const float* __restrict__ bias,
                               float* __restrict__ out, int ldo,
                               size_t part_stride, int n_groups, int K) {
  const int unit = (blockIdx.x * blockDim.x + threadIdx.x) >> 5;
  const int lane = threadIdx.x & 31;
  if (unit >= n_groups * KSPLIT) return;  // wave-uniform: EXEC all-1s at WMMA

  const int ks = unit % KSPLIT;
  const int ng = unit / KSPLIT;
  const int klen  = K / KSPLIT;
  const int kbase = ks * klen;

  const int half = lane >> 4;
  const int l15  = lane & 15;

  const bf16_t* __restrict__ arow0 = A + (size_t)l15 * lda + kbase;
  const bf16_t* __restrict__ arow1 = A + (size_t)(16 + l15) * lda + kbase;
  const bf16_t* __restrict__ brow[TILES];
#pragma unroll
  for (int j = 0; j < TILES; ++j)
    brow[j] = B + (size_t)(ng * (TILES * 16) + j * 16 + l15) * ldb + kbase;

  v8f acc[2][TILES] = {};

  auto load_all = [&](int k) {
    Frags<TILES> f;
    f.a0 = load_frag(arow0, k, half);
    f.a1 = load_frag(arow1, k, half);
#pragma unroll
    for (int j = 0; j < TILES; ++j) f.b[j] = load_frag(brow[j], k, half);
    return f;
  };
  auto compute = [&](const Frags<TILES>& f) {
#pragma unroll
    for (int j = 0; j < TILES; ++j) {
      WMMA_BF16(acc[0][j], f.a0, f.b[j]);
      WMMA_BF16(acc[1][j], f.a1, f.b[j]);
    }
  };

  // software pipeline: loads for step k are in flight while step k-32 computes
  Frags<TILES> cur = load_all(0);
#pragma unroll 2
  for (int k = 32; k < klen; k += 32) {
    Frags<TILES> nxt = load_all(k);
    compute(cur);
    cur = nxt;
  }
  compute(cur);

  // C/D layout: VGPR r, lanes 0-15 -> M=r, N=lane; lanes 16-31 -> M=8+r.
  float* outp = out + (size_t)ks * part_stride;
#pragma unroll
  for (int j = 0; j < TILES; ++j) {
    const int col = ng * (TILES * 16) + j * 16 + l15;
    const float bv = bias ? bias[col] : 0.0f;
    float* op = outp + col;
#pragma unroll
    for (int mt = 0; mt < 2; ++mt) {
#pragma unroll
      for (int r = 0; r < 8; ++r) {
        const int row = mt * 16 + half * 8 + r;
        op[(size_t)row * ldo] = acc[mt][j][r] + bv;
      }
    }
  }
}

// Convert W_ih|W_hh to one concatenated bf16 matrix (G_ x KCAT_); sum biases.
__global__ void prep_gate_weights(const float* __restrict__ W_ih,
                                  const float* __restrict__ W_hh,
                                  const float* __restrict__ b_ih,
                                  const float* __restrict__ b_hh,
                                  bf16_t* __restrict__ Wcat,
                                  float* __restrict__ bsum) {
  const size_t total = (size_t)G_ * KCAT_;
  for (size_t idx = (size_t)blockIdx.x * blockDim.x + threadIdx.x; idx < total;
       idx += (size_t)gridDim.x * blockDim.x) {
    const int row = (int)(idx >> 11);          // / 2048
    const int col = (int)(idx & 2047);
    const float v = (col < E_) ? W_ih[(size_t)row * E_ + col]
                               : W_hh[(size_t)row * H_ + (col - E_)];
    Wcat[idx] = (bf16_t)v;
    if (idx < (size_t)G_) bsum[idx] = b_ih[idx] + b_hh[idx];
  }
}

__global__ void prep_emit_weights(const float* __restrict__ emit_W,
                                  bf16_t* __restrict__ Wemit) {
  const size_t total = (size_t)V_ * H_;
  for (size_t idx = (size_t)blockIdx.x * blockDim.x + threadIdx.x; idx < total;
       idx += (size_t)gridDim.x * blockDim.x)
    Wemit[idx] = (bf16_t)emit_W[idx];
}

// c <- c0 ; xh[:, E_:] <- bf16(h0)
__global__ void init_state(const float* __restrict__ h0,
                           const float* __restrict__ c0,
                           bf16_t* __restrict__ xh,
                           float* __restrict__ c) {
  const int idx = blockIdx.x * blockDim.x + threadIdx.x;   // N_*H_
  const int n = idx >> 10, j = idx & 1023;
  c[idx] = c0[idx];
  xh[(size_t)n * KCAT_ + E_ + j] = (bf16_t)h0[idx];
}

// xh[:, :E_] <- bf16(embed_W[words[:, t]])
__global__ void embed_step(const int* __restrict__ words,
                           const float* __restrict__ embed_W,
                           bf16_t* __restrict__ xh, int t) {
  const int idx = blockIdx.x * blockDim.x + threadIdx.x;   // N_*E_
  const int n = idx >> 10, e = idx & 1023;
  const int w = words[n * T_ + t];
  xh[(size_t)n * KCAT_ + e] = (bf16_t)embed_W[(size_t)w * E_ + e];
}

__device__ __forceinline__ float sigmoidf_(float x) {
  return 1.0f / (1.0f + __expf(-x));
}

// LSTM pointwise cell: gates = g0 + g1 + bsum (split-K reduction + bias),
// then updates c and writes bf16(h) into xh[:, E_:].
__global__ void lstm_cell(const float* __restrict__ g0,
                          const float* __restrict__ g1,
                          const float* __restrict__ bsum,
                          float* __restrict__ c,
                          bf16_t* __restrict__ xh) {
  const int idx = blockIdx.x * blockDim.x + threadIdx.x;   // N_*H_
  const int n = idx >> 10, j = idx & 1023;
  const size_t base = (size_t)n * G_;
  const float iv = g0[base + j]          + g1[base + j]          + bsum[j];
  const float fv = g0[base + H_ + j]     + g1[base + H_ + j]     + bsum[H_ + j];
  const float gv = g0[base + 2 * H_ + j] + g1[base + 2 * H_ + j] + bsum[2 * H_ + j];
  const float ov = g0[base + 3 * H_ + j] + g1[base + 3 * H_ + j] + bsum[3 * H_ + j];
  const float ig = sigmoidf_(iv);
  const float fg = sigmoidf_(fv);
  const float gg = tanhf(gv);
  const float og = sigmoidf_(ov);
  const float cn = fg * c[idx] + ig * gg;
  c[idx] = cn;
  const float hn = og * tanhf(cn);
  xh[(size_t)n * KCAT_ + E_ + j] = (bf16_t)hn;
}

// Per-row online log-softmax + target gather. One block per batch row.
__global__ void loss_step(const float* __restrict__ logits,
                          const int* __restrict__ words,
                          float* __restrict__ out, int t) {
  __shared__ float sm[256];
  __shared__ float ss[256];
  const int n = blockIdx.x;
  const float* __restrict__ row = logits + (size_t)n * V_;
  float m = -INFINITY, s = 0.0f;
  for (int j = threadIdx.x; j < V_; j += blockDim.x) {
    const float x = row[j];
    const float nm = fmaxf(m, x);
    s = s * __expf(m - nm) + __expf(x - nm);
    m = nm;
  }
  sm[threadIdx.x] = m;
  ss[threadIdx.x] = s;
  __syncthreads();
  for (int off = 128; off > 0; off >>= 1) {
    if (threadIdx.x < off) {
      const float m1 = sm[threadIdx.x], s1 = ss[threadIdx.x];
      const float m2 = sm[threadIdx.x + off], s2 = ss[threadIdx.x + off];
      const float nm = fmaxf(m1, m2);
      ss[threadIdx.x] = s1 * __expf(m1 - nm) + s2 * __expf(m2 - nm);
      sm[threadIdx.x] = nm;
    }
    __syncthreads();
  }
  if (threadIdx.x == 0) {
    const int tgt = words[n * T_ + t + 1];
    out[n * (T_ - 1) + t] = row[tgt] - sm[0] - __logf(ss[0]);
  }
}

extern "C" void kernel_launch(void* const* d_in, const int* in_sizes, int n_in,
                              void* d_out, int out_size, void* d_ws,
                              size_t ws_size, hipStream_t stream) {
  (void)in_sizes; (void)n_in; (void)out_size; (void)ws_size;
  const int*   words   = (const int*)  d_in[0];
  const float* h0      = (const float*)d_in[1];
  const float* c0      = (const float*)d_in[2];
  const float* embed_W = (const float*)d_in[3];
  const float* W_ih    = (const float*)d_in[4];
  const float* W_hh    = (const float*)d_in[5];
  const float* b_ih    = (const float*)d_in[6];
  const float* b_hh    = (const float*)d_in[7];
  const float* emit_W  = (const float*)d_in[8];
  const float* emit_b  = (const float*)d_in[9];
  float* out = (float*)d_out;

  // workspace layout (all chunks are multiples of 256 B)
  char* ws = (char*)d_ws;
  constexpr size_t SZ_WCAT  = (size_t)G_ * KCAT_ * sizeof(bf16_t); // 16 MB
  constexpr size_t SZ_WEMIT = (size_t)V_ * H_   * sizeof(bf16_t);  // 65.5 MB
  constexpr size_t SZ_BSUM  = (size_t)G_ * sizeof(float);
  constexpr size_t SZ_XH    = (size_t)N_ * KCAT_ * sizeof(bf16_t);
  constexpr size_t SZ_C     = (size_t)N_ * H_ * sizeof(float);
  constexpr size_t SZ_GATES = (size_t)N_ * G_ * sizeof(float);     // per K-half
  size_t off = 0;
  bf16_t* Wcat   = (bf16_t*)(ws + off); off += SZ_WCAT;
  bf16_t* Wemit  = (bf16_t*)(ws + off); off += SZ_WEMIT;
  float*  bsum   = (float*) (ws + off); off += SZ_BSUM;
  bf16_t* xh     = (bf16_t*)(ws + off); off += SZ_XH;
  float*  cbuf   = (float*) (ws + off); off += SZ_C;
  float*  gates  = (float*) (ws + off); off += 2 * SZ_GATES;       // 2 partials
  float*  logits = (float*) (ws + off);

  // one-time (per call) weight downconversion — amortized over 63 steps
  prep_gate_weights<<<4096, 256, 0, stream>>>(W_ih, W_hh, b_ih, b_hh, Wcat, bsum);
  prep_emit_weights<<<8192, 256, 0, stream>>>(emit_W, Wemit);
  init_state<<<(N_ * H_) / 256, 256, 0, stream>>>(h0, c0, xh, cbuf);

  // gates: 64 strips of 32x64, split-K=2 -> 128 wave-units -> 16 blocks
  constexpr int NG_GATES = G_ / 64;
  constexpr int UNITS_GATES = NG_GATES * 2;
  const int blk_gates = (UNITS_GATES + 7) / 8;
  // emit: 1000 strips of 32x32 -> 1000 wave-units -> 125 blocks
  constexpr int NG_EMIT = V_ / 32;
  const int blk_emit = (NG_EMIT + 7) / 8;

  const size_t gate_part = (size_t)N_ * G_;   // elements per K-partial

  for (int t = 0; t < T_ - 1; ++t) {
    embed_step<<<(N_ * E_) / 256, 256, 0, stream>>>(words, embed_W, xh, t);
    gemm_bf16_wmma<4, 2><<<blk_gates, 256, 0, stream>>>(
        xh, KCAT_, Wcat, KCAT_, nullptr, gates, G_, gate_part, NG_GATES, KCAT_);
    lstm_cell<<<(N_ * H_) / 256, 256, 0, stream>>>(gates, gates + gate_part,
                                                   bsum, cbuf, xh);
    gemm_bf16_wmma<2, 1><<<blk_emit, 256, 0, stream>>>(
        xh + E_, KCAT_, Wemit, H_, emit_b, logits, V_, 0, NG_EMIT, H_);
    loss_step<<<N_, 256, 0, stream>>>(logits, words, out, t);
  }
}
